// LSTMLayerCustom_83786222010378
// MI455X (gfx1250) — matrix operand order repaired and checked by reading, not compiled, not run on hardware
//
#include <hip/hip_runtime.h>
#include <hip/hip_bf16.h>

// ---------------------------------------------------------------------------
// Bidirectional 2-layer LSTM for MI455X (gfx1250), wave32 + WMMA bf16.
//   gates = [x_t, h_{t-1}] @ [Wih; Whh]^T + (bih+bhh)    (f32 accumulate)
//   v_wmma_f32_16x16x32_bf16 GEMM; LSTM cell elementwise fused in-wave.
// Each wave: 2 M-tiles (32 batch rows) x 16 hidden cols x all 4 gate groups,
// so B-weight fragments are reused across both M-tiles (halves L2 traffic).
// ---------------------------------------------------------------------------

typedef __attribute__((ext_vector_type(16))) __bf16 v16bf;
typedef __attribute__((ext_vector_type(8)))  float  v8f;

struct __align__(16) U4 { unsigned int x, y, z, w; };
union FragBF { v16bf v; U4 q[2]; };

#define DEVINL __device__ __forceinline__

constexpr int BB = 64;    // batch
constexpr int TT = 256;   // time
constexpr int II = 512;   // input dim
constexpr int HH = 1024;  // hidden dim
constexpr int GG = 4096;  // 4*H (gates)
constexpr int LL = 2;     // layers

// ---------------- workspace layout (bytes) ----------------
constexpr size_t OFF_XBF  = 0;                                    // bf16 [T][B][I]
constexpr size_t SZ_XBF   = (size_t)TT * BB * II * 2;
constexpr size_t OFF_WIH0 = OFF_XBF + SZ_XBF;                     // bf16 packed [dir][G*I]
constexpr size_t SZ_WIH0  = (size_t)2 * GG * II * 2;
constexpr size_t OFF_WHH0 = OFF_WIH0 + SZ_WIH0;                   // bf16 packed [dir][G*H]
constexpr size_t SZ_WHH0  = (size_t)2 * GG * HH * 2;
constexpr size_t OFF_WIH1 = OFF_WHH0 + SZ_WHH0;
constexpr size_t SZ_WIH1  = SZ_WHH0;
constexpr size_t OFF_WHH1 = OFF_WIH1 + SZ_WIH1;
constexpr size_t SZ_WHH1  = SZ_WHH0;
constexpr size_t OFF_BIAS = OFF_WHH1 + SZ_WHH1;                   // f32 [dir][l][G]
constexpr size_t SZ_BIAS  = (size_t)2 * LL * GG * 4;
constexpr size_t OFF_H    = OFF_BIAS + SZ_BIAS;                   // bf16 [dir][l][parity][B][H]
constexpr size_t SZ_H     = (size_t)2 * LL * 2 * BB * HH * 2;
constexpr size_t OFF_C    = OFF_H + SZ_H;                         // f32 [dir][l][B][H]
constexpr size_t SZ_C     = (size_t)2 * LL * BB * HH * 4;

// ---------------- helpers ----------------
DEVINL unsigned short f2bf(float f) {
    unsigned int u = __float_as_uint(f);
    unsigned int r = u + 0x7FFFu + ((u >> 16) & 1u);   // round-to-nearest-even
    return (unsigned short)(r >> 16);
}
DEVINL float sigm(float x) { return 1.0f / (1.0f + __expf(-x)); }
DEVINL float tanh_fast(float x) { return 2.0f / (1.0f + __expf(-2.0f * x)) - 1.0f; }

// ---------------- one-time prep kernels ----------------

// zero h double-buffers and c working state (must run every launch: determinism)
__global__ void init_state_kernel(char* __restrict__ ws) {
    size_t idx = (size_t)blockIdx.x * blockDim.x + threadIdx.x;
    size_t words = (SZ_H + SZ_C) / 4;
    if (idx < words) ((unsigned int*)(ws + OFF_H))[idx] = 0u;
}

// x[b][t][i] (f32) -> xbf[t][b][i] (bf16)
__global__ void convert_x_kernel(const float* __restrict__ x, unsigned short* __restrict__ xbf) {
    size_t idx = (size_t)blockIdx.x * blockDim.x + threadIdx.x;
    if (idx >= (size_t)TT * BB * II) return;
    int i = (int)(idx % II);
    size_t bt = idx / II;
    int b = (int)(bt % BB);
    int t = (int)(bt / BB);
    xbf[idx] = f2bf(x[((size_t)b * TT + t) * II + i]);
}

// Pack W (row-major [4096][Kdim], gates = act @ W^T) into WMMA B-fragment tiles:
// tile (nt,kt) covers N=nt*16..+15, K=kt*32..+31; element (lane,e):
//   n = nt*16 + (lane&15), k = kt*32 + (e>>3)*16 + (lane>>4)*8 + (e&7)
// flat dst index = ((nt*KT + kt)*32 + lane)*16 + e  -> each lane reads 32B contiguous.
__global__ void pack_weights_kernel(const float* __restrict__ W,
                                    unsigned short* __restrict__ dst, int Kdim) {
    size_t idx = (size_t)blockIdx.x * blockDim.x + threadIdx.x;
    size_t total = (size_t)GG * Kdim;
    if (idx >= total) return;
    int e    = (int)(idx & 15);
    int lane = (int)((idx >> 4) & 31);
    size_t tile = idx >> 9;
    int KT = Kdim >> 5;
    int kt = (int)(tile % KT);
    int nt = (int)(tile / KT);
    int n  = nt * 16 + (lane & 15);
    int k  = kt * 32 + ((e >> 3) << 4) + ((lane >> 4) << 3) + (e & 7);
    dst[idx] = f2bf(W[(size_t)n * Kdim + k]);
}

// bias[dir][l][G] = bih + bhh
__global__ void bias_kernel(const float* __restrict__ bf0, const float* __restrict__ bhf0,
                            const float* __restrict__ bf1, const float* __restrict__ bhf1,
                            const float* __restrict__ bb0, const float* __restrict__ bhb0,
                            const float* __restrict__ bb1, const float* __restrict__ bhb1,
                            float* __restrict__ biasOut) {
    int idx = blockIdx.x * blockDim.x + threadIdx.x;
    if (idx >= 2 * LL * GG) return;
    int j  = idx % GG;
    int dl = idx / GG;  // dir*2 + layer
    const float* a; const float* b;
    if      (dl == 0) { a = bf0; b = bhf0; }
    else if (dl == 1) { a = bf1; b = bhf1; }
    else if (dl == 2) { a = bb0; b = bhb0; }
    else              { a = bb1; b = bhb1; }
    biasOut[idx] = a[j] + b[j];
}

// ---------------- fused step kernel ----------------
// One wave: 2 M-tiles (mp -> batch rows 32*mp..+31), N-slice = 16 hidden cols
// (nc), all 4 gate groups. acc[u*4+gi] is the 16x16 f32 tile for M-tile u,
// gate gi. B fragment loaded once per (kt,gi), used by both M-tiles.

DEVINL void gemm_acc(const unsigned short* __restrict__ A, int K,
                     const unsigned short* __restrict__ W,
                     v8f acc[8], int mp, int nc, int lane) {
    const int KT = K >> 5;
    // A 16x32 bf16 layout: lane m=lane&15 (half=lane>>4), two contiguous 8-elem runs.
    const unsigned short* arow0 =
        A + (size_t)(mp * 32 + (lane & 15)) * K + ((lane >> 4) * 8);
    const unsigned short* arow1 = arow0 + (size_t)16 * K;
    for (int kt = 0; kt < KT; ++kt) {
        FragBF a0, a1;
        a0.q[0] = *(const U4*)(arow0 + kt * 32);
        a0.q[1] = *(const U4*)(arow0 + kt * 32 + 16);
        a1.q[0] = *(const U4*)(arow1 + kt * 32);
        a1.q[1] = *(const U4*)(arow1 + kt * 32 + 16);
#pragma unroll
        for (int gi = 0; gi < 4; ++gi) {
            const U4* wp = (const U4*)(W +
                ((size_t)((gi * 64 + nc) * KT + kt) * 32 + lane) * 16);
            FragBF b;
            b.q[0] = wp[0];
            b.q[1] = wp[1];
            acc[gi] = __builtin_amdgcn_wmma_f32_16x16x32_bf16(
                false, a0.v, false, b.v, (short)0, acc[gi], false, false);
            acc[4 + gi] = __builtin_amdgcn_wmma_f32_16x16x32_bf16(
                false, a1.v, false, b.v, (short)0, acc[4 + gi], false, false);
        }
    }
}

__global__ void __launch_bounds__(32)
lstm_step_kernel(char* __restrict__ ws, float* __restrict__ out, int s, int layer) {
    const int nc   = blockIdx.x;   // hidden chunk: 16 cols
    const int mp   = blockIdx.y;   // batch pair-tile: 32 rows
    const int dir  = blockIdx.z;   // 0 = fwd, 1 = bwd
    const int lane = threadIdx.x;  // wave32
    const int p    = s & 1;
    const int t    = dir ? (TT - 1 - s) : s;

    unsigned short* hbase = (unsigned short*)(ws + OFF_H);
    auto hbuf = [&](int d, int l, int par) {
        return hbase + (size_t)((d * LL + l) * 2 + par) * BB * HH;
    };

    const unsigned short* Ain;
    const unsigned short* Wih;
    int Kin;
    if (layer == 0) {
        Ain = (const unsigned short*)(ws + OFF_XBF) + (size_t)t * BB * II;
        Wih = (const unsigned short*)(ws + OFF_WIH0) + (size_t)dir * GG * II;
        Kin = II;
    } else {
        Ain = hbuf(dir, 0, 1 - p);  // layer0 output of this same step
        Wih = (const unsigned short*)(ws + OFF_WIH1) + (size_t)dir * GG * HH;
        Kin = HH;
    }
    const unsigned short* Ah =
        hbuf(dir, layer, p);        // h_{t-1} (parity-double-buffered: race-free)
    const unsigned short* Whh =
        (const unsigned short*)(ws + (layer ? OFF_WHH1 : OFF_WHH0)) + (size_t)dir * GG * HH;
    const float* bias = (const float*)(ws + OFF_BIAS) + (size_t)(dir * LL + layer) * GG;
    float* cw = (float*)(ws + OFF_C) + (size_t)(dir * LL + layer) * BB * HH;
    unsigned short* hnew = hbuf(dir, layer, 1 - p);

    v8f acc[8];
#pragma unroll
    for (int q = 0; q < 8; ++q)
        acc[q] = v8f{0.f, 0.f, 0.f, 0.f, 0.f, 0.f, 0.f, 0.f};

    gemm_acc(Ain, Kin, Wih, acc, mp, nc, lane);   // input contribution
    gemm_acc(Ah,  HH,  Whh, acc, mp, nc, lane);   // recurrent contribution

    // C/D layout: vgpr r, lanes 0-15 -> M=r, lanes 16-31 -> M=r+8; N=lane&15
    const int n    = lane & 15;
    const int mh   = lane >> 4;
    const int hidx = nc * 16 + n;
    const float bi = bias[hidx];
    const float bf = bias[HH + hidx];
    const float bg = bias[2 * HH + hidx];
    const float bo = bias[3 * HH + hidx];

#pragma unroll
    for (int u = 0; u < 2; ++u) {
#pragma unroll
        for (int r = 0; r < 8; ++r) {
            int b = mp * 32 + u * 16 + mh * 8 + r;
            float iv = acc[u * 4 + 0][r] + bi;
            float fv = acc[u * 4 + 1][r] + bf;
            float gv = acc[u * 4 + 2][r] + bg;
            float ov = acc[u * 4 + 3][r] + bo;
            float cp = cw[(size_t)b * HH + hidx];
            float cn = sigm(fv) * cp + sigm(iv) * tanh_fast(gv);
            float hn = sigm(ov) * tanh_fast(cn);
            cw[(size_t)b * HH + hidx]   = cn;
            hnew[(size_t)b * HH + hidx] = f2bf(hn);
            size_t ho = (((size_t)t * LL + layer) * BB + b) * (2 * HH)
                        + (size_t)dir * HH + hidx;
            out[ho] = hn;                                   // hs
            out[(size_t)TT * LL * BB * 2 * HH + ho] = cn;   // cs
        }
    }
}

// ---------------- host launcher ----------------
extern "C" void kernel_launch(void* const* d_in, const int* in_sizes, int n_in,
                              void* d_out, int out_size, void* d_ws, size_t ws_size,
                              hipStream_t stream) {
    char*  ws  = (char*)d_ws;
    float* out = (float*)d_out;

    const float* x = (const float*)d_in[0];
    // input order: x, then f0(Wih,Whh,bih,bhh), f1(...), b0(...), b1(...)
    const float* Wih_f0 = (const float*)d_in[1];
    const float* Whh_f0 = (const float*)d_in[2];
    const float* bih_f0 = (const float*)d_in[3];
    const float* bhh_f0 = (const float*)d_in[4];
    const float* Wih_f1 = (const float*)d_in[5];
    const float* Whh_f1 = (const float*)d_in[6];
    const float* bih_f1 = (const float*)d_in[7];
    const float* bhh_f1 = (const float*)d_in[8];
    const float* Wih_b0 = (const float*)d_in[9];
    const float* Whh_b0 = (const float*)d_in[10];
    const float* bih_b0 = (const float*)d_in[11];
    const float* bhh_b0 = (const float*)d_in[12];
    const float* Wih_b1 = (const float*)d_in[13];
    const float* Whh_b1 = (const float*)d_in[14];
    const float* bih_b1 = (const float*)d_in[15];
    const float* bhh_b1 = (const float*)d_in[16];

    unsigned short* wih0 = (unsigned short*)(ws + OFF_WIH0);
    unsigned short* whh0 = (unsigned short*)(ws + OFF_WHH0);
    unsigned short* wih1 = (unsigned short*)(ws + OFF_WIH1);
    unsigned short* whh1 = (unsigned short*)(ws + OFF_WHH1);
    float*          bias = (float*)(ws + OFF_BIAS);
    unsigned short* xbf  = (unsigned short*)(ws + OFF_XBF);

    // state init (h parity buffers + c) — every call, for determinism
    {
        size_t words = (SZ_H + SZ_C) / 4;
        init_state_kernel<<<(unsigned)((words + 255) / 256), 256, 0, stream>>>(ws);
    }
    // x -> bf16 [t][b][i]
    {
        size_t total = (size_t)TT * BB * II;
        convert_x_kernel<<<(unsigned)((total + 255) / 256), 256, 0, stream>>>(x, xbf);
    }
    // pack weights into WMMA B-fragment tiles (bf16), dir-major
    {
        size_t tI = (size_t)GG * II, tH = (size_t)GG * HH;
        unsigned gI = (unsigned)((tI + 255) / 256), gH = (unsigned)((tH + 255) / 256);
        pack_weights_kernel<<<gI, 256, 0, stream>>>(Wih_f0, wih0,            II);
        pack_weights_kernel<<<gI, 256, 0, stream>>>(Wih_b0, wih0 + tI,       II);
        pack_weights_kernel<<<gH, 256, 0, stream>>>(Whh_f0, whh0,            HH);
        pack_weights_kernel<<<gH, 256, 0, stream>>>(Whh_b0, whh0 + tH,       HH);
        pack_weights_kernel<<<gH, 256, 0, stream>>>(Wih_f1, wih1,            HH);
        pack_weights_kernel<<<gH, 256, 0, stream>>>(Wih_b1, wih1 + tH,       HH);
        pack_weights_kernel<<<gH, 256, 0, stream>>>(Whh_f1, whh1,            HH);
        pack_weights_kernel<<<gH, 256, 0, stream>>>(Whh_b1, whh1 + tH,       HH);
    }
    // combined biases
    bias_kernel<<<(2 * LL * GG + 255) / 256, 256, 0, stream>>>(
        bih_f0, bhh_f0, bih_f1, bhh_f1, bih_b0, bhh_b0, bih_b1, bhh_b1, bias);

    // recurrence: 256 steps x 2 layers; fwd+bwd fused via grid.z
    dim3 grid(HH / 16, BB / 32, 2);  // 64 x 2 x 2 waves, 2 M-tiles per wave
    dim3 block(32, 1, 1);
    for (int s = 0; s < TT; ++s) {
        lstm_step_kernel<<<grid, block, 0, stream>>>(ws, out, s, 0);
        lstm_step_kernel<<<grid, block, 0, stream>>>(ws, out, s, 1);
    }
}
